// SelfAttention_29480655520444
// MI455X (gfx1250) — matrix operand order repaired and checked by reading, not compiled, not run on hardware
//
#include <hip/hip_runtime.h>
#include <hip/hip_bf16.h>

// ---------------------------------------------------------------------------
// SAGAN self-attention, B=4 C=256 H=W=64 (N=4096, D=32), fp32 in/out.
// softmax(axis=-2) ==> flash attention with Q<-Wk proj, K<-Wq proj, V<-Wv proj.
// Everything matmul-shaped (projections AND attention) runs on
// v_wmma_f32_16x16x32_bf16 with fp32 accumulation; softmax stats in fp32.
// Transposed dataflow: S^T = K*Q^T and O^T = V^T*P^T so the softmax axis (n)
// falls on VGPR slots and softmax stats are per-lane for the O accumulator.
// m-tile = 32 per block: each V fragment feeds 4 WMMAs (halves L2 V traffic).
// ---------------------------------------------------------------------------

#define B_DIM 4
#define C_DIM 256
#define N_DIM 4096
#define D_DIM 32
#define PSTRIDE 264   // 256 + 8 halfs pad -> b128 reads stripe across banks

typedef __attribute__((ext_vector_type(16))) __bf16 v16bf;
typedef __attribute__((ext_vector_type(8)))  __bf16 v8bf;
typedef __attribute__((ext_vector_type(8)))  float  v8f;

union BF16Frag { v16bf v; v8bf h[2]; };

static __device__ __forceinline__ v8f wmma_bf16(v16bf a, v16bf b, v8f c) {
    return __builtin_amdgcn_wmma_f32_16x16x32_bf16(false, a, false, b,
                                                   (short)0, c, false, false);
}

// ---------------------------------------------------------------------------
// Kernel 0a: x [B][C][N] fp32 -> xT [B][N][C] bf16 via 64x64 LDS transpose.
// ---------------------------------------------------------------------------
__global__ __launch_bounds__(256) void transpose_x_kernel(
    const float* __restrict__ x, __bf16* __restrict__ xT)
{
    __shared__ float tile[64][65];
    const int b  = blockIdx.x >> 8;              // 4*64 tiles per batch
    const int ct = ((blockIdx.x >> 6) & 3) * 64;
    const int nt = (blockIdx.x & 63) * 64;
    const int u  = threadIdx.x & 63;
    const int g  = threadIdx.x >> 6;

    #pragma unroll 4
    for (int i = 0; i < 16; ++i) {
        const int c = i * 4 + g;
        tile[c][u] = x[(size_t)(b * C_DIM + ct + c) * N_DIM + nt + u];
    }
    __syncthreads();
    #pragma unroll 4
    for (int i = 0; i < 16; ++i) {
        const int n = i * 4 + g;
        xT[(size_t)(b * N_DIM + nt + n) * C_DIM + ct + u] = (__bf16)tile[u][n];
    }
}

// ---------------------------------------------------------------------------
// Kernel 0b: convert weights to bf16 (one-time, tiny).
// ---------------------------------------------------------------------------
__global__ __launch_bounds__(256) void convert_w_kernel(
    const float* __restrict__ Wq, const float* __restrict__ Wk,
    const float* __restrict__ Wv,
    __bf16* __restrict__ Wqb, __bf16* __restrict__ Wkb,
    __bf16* __restrict__ Wvb)
{
    const int i = blockIdx.x * 256 + threadIdx.x;       // 65536 threads
    if (i < D_DIM * C_DIM) {
        Wqb[i] = (__bf16)Wq[i];
        Wkb[i] = (__bf16)Wk[i];
    }
    Wvb[i] = (__bf16)Wv[i];                             // 256*256 exactly
}

// ---------------------------------------------------------------------------
// Kernel 1a: q/k projections on WMMA.
//   attnK[b][n][d] = Wq.x + bq (keys), attnQ[b][n][d] = Wk.x + bk (queries)
// D-tile: rows n (VGPR), cols d (lane). A = xT rows n (k=c contiguous),
// B = W^T cols d (k=c contiguous = W row-major). Wave owns one 16-n tile.
// ---------------------------------------------------------------------------
__global__ __launch_bounds__(256) void proj_qk_kernel(
    const __bf16* __restrict__ xT,
    const __bf16* __restrict__ Wqb, const float* __restrict__ bq,
    const __bf16* __restrict__ Wkb, const float* __restrict__ bk,
    __bf16* __restrict__ attnK, __bf16* __restrict__ attnQ)
{
    const int b    = blockIdx.x >> 5;                       // N/128 chunks
    const int nt   = ((blockIdx.x & 31) * 8 + (threadIdx.x >> 5)) * 16;
    const int lane = threadIdx.x & 31;
    const int half = lane >> 4, l16 = lane & 15;
    const int d0   = half * 16;

    v8f accK0 = {}, accK1 = {}, accQ0 = {}, accQ1 = {};
    const __bf16* xrow = xT + (size_t)(b * N_DIM + nt + l16) * C_DIM;
    const __bf16* qr0  = Wqb + (size_t)l16 * C_DIM;         // B col d = l16
    const __bf16* qr1  = qr0 + 16 * C_DIM;                  // B col d = l16+16
    const __bf16* kr0  = Wkb + (size_t)l16 * C_DIM;
    const __bf16* kr1  = kr0 + 16 * C_DIM;

    for (int cc = 0; cc < C_DIM; cc += 32) {
        BF16Frag xa;                                        // A frag
        xa.h[0] = *(const v8bf*)(xrow + cc + d0);
        xa.h[1] = *(const v8bf*)(xrow + cc + d0 + 8);
        BF16Frag wq0, wq1, wk0, wk1;                        // B frags
        wq0.h[0] = *(const v8bf*)(qr0 + cc + half * 16);
        wq0.h[1] = *(const v8bf*)(qr0 + cc + half * 16 + 8);
        wq1.h[0] = *(const v8bf*)(qr1 + cc + half * 16);
        wq1.h[1] = *(const v8bf*)(qr1 + cc + half * 16 + 8);
        wk0.h[0] = *(const v8bf*)(kr0 + cc + half * 16);
        wk0.h[1] = *(const v8bf*)(kr0 + cc + half * 16 + 8);
        wk1.h[0] = *(const v8bf*)(kr1 + cc + half * 16);
        wk1.h[1] = *(const v8bf*)(kr1 + cc + half * 16 + 8);
        accK0 = wmma_bf16(xa.v, wq0.v, accK0);              // keys from Wq
        accK1 = wmma_bf16(xa.v, wq1.v, accK1);
        accQ0 = wmma_bf16(xa.v, wk0.v, accQ0);              // queries from Wk
        accQ1 = wmma_bf16(xa.v, wk1.v, accQ1);
    }
    const float bK0 = bq[l16], bK1 = bq[16 + l16];
    const float bQ0 = bk[l16], bQ1 = bk[16 + l16];
    #pragma unroll
    for (int j = 0; j < 8; ++j) {
        const size_t row = (size_t)(b * N_DIM + nt + j + half * 8) * D_DIM;
        attnK[row + l16]      = (__bf16)(accK0[j] + bK0);
        attnK[row + 16 + l16] = (__bf16)(accK1[j] + bK1);
        attnQ[row + l16]      = (__bf16)(accQ0[j] + bQ0);
        attnQ[row + 16 + l16] = (__bf16)(accQ1[j] + bQ1);
    }
}

// ---------------------------------------------------------------------------
// Kernel 1b: v projection on WMMA.  attnV[b][c][n] = Wv.x + bv
// D-tile: rows c_out (VGPR), cols n (lane). A = Wv rows c_out (k=c_in
// contiguous), B = xT cols n (k=c_in contiguous). 2x2 register tile/wave.
// ---------------------------------------------------------------------------
__global__ __launch_bounds__(256) void proj_v_kernel(
    const __bf16* __restrict__ xT, const __bf16* __restrict__ Wvb,
    const float* __restrict__ bv, __bf16* __restrict__ attnV)
{
    const int b    = blockIdx.x >> 7;            // N/32 chunks per batch
    const int nt   = (blockIdx.x & 127) * 32;
    const int wave = threadIdx.x >> 5;
    const int lane = threadIdx.x & 31;
    const int half = lane >> 4, l16 = lane & 15;
    const int c0   = wave * 32;                  // wave's c_out slice
    const int d0   = half * 16;

    v8f a00 = {}, a01 = {}, a10 = {}, a11 = {};  // [c-tile][n-tile]
    const __bf16* wrow0 = Wvb + (size_t)(c0 + l16) * C_DIM;
    const __bf16* wrow1 = wrow0 + 16 * C_DIM;
    const __bf16* xrow0 = xT + (size_t)(b * N_DIM + nt + l16) * C_DIM;
    const __bf16* xrow1 = xrow0 + 16 * C_DIM;

    for (int cc = 0; cc < C_DIM; cc += 32) {
        BF16Frag wa0, wa1;                       // A frags: Wv rows c_out
        wa0.h[0] = *(const v8bf*)(wrow0 + cc + d0);
        wa0.h[1] = *(const v8bf*)(wrow0 + cc + d0 + 8);
        wa1.h[0] = *(const v8bf*)(wrow1 + cc + d0);
        wa1.h[1] = *(const v8bf*)(wrow1 + cc + d0 + 8);
        BF16Frag xb0, xb1;                       // B frags: xT cols n
        xb0.h[0] = *(const v8bf*)(xrow0 + cc + half * 16);
        xb0.h[1] = *(const v8bf*)(xrow0 + cc + half * 16 + 8);
        xb1.h[0] = *(const v8bf*)(xrow1 + cc + half * 16);
        xb1.h[1] = *(const v8bf*)(xrow1 + cc + half * 16 + 8);
        a00 = wmma_bf16(wa0.v, xb0.v, a00);
        a01 = wmma_bf16(wa0.v, xb1.v, a01);
        a10 = wmma_bf16(wa1.v, xb0.v, a10);
        a11 = wmma_bf16(wa1.v, xb1.v, a11);
    }
    #pragma unroll
    for (int j = 0; j < 8; ++j) {
        const int cA = c0 + j + half * 8;
        const int cB = cA + 16;
        const float bA = bv[cA], bB = bv[cB];
        __bf16* rA = attnV + (size_t)(b * C_DIM + cA) * N_DIM + nt;
        __bf16* rB = attnV + (size_t)(b * C_DIM + cB) * N_DIM + nt;
        rA[l16]      = (__bf16)(a00[j] + bA);
        rA[16 + l16] = (__bf16)(a01[j] + bA);
        rB[l16]      = (__bf16)(a10[j] + bB);
        rB[16 + l16] = (__bf16)(a11[j] + bB);
    }
}

// ---------------------------------------------------------------------------
// Kernel 2: cooperative flash attention. Grid: B*N/32 blocks x 8 waves.
// Block owns a 32-wide m-tile (2 subtiles); super-iteration covers 256 keys:
//   pass A: wave w computes S^T for keys [ns+32w, ns+32w+32)   (4 WMMA)
//   pass B: global max via LDS, exp, P^T -> shared LDS, partial sums
//   pass C: every wave runs P.V for its 32-channel slice       (32 WMMA,
//           each V fragment pair feeds 4 WMMAs -> half the V L2 traffic)
// ---------------------------------------------------------------------------
__global__ __launch_bounds__(256) void sagan_flash_attn_kernel(
    const float* __restrict__ x, const float* __restrict__ gammaPtr,
    const __bf16* __restrict__ attnQ, const __bf16* __restrict__ attnK,
    const __bf16* __restrict__ attnV, float* __restrict__ out)
{
    __shared__ __align__(16) __bf16 ldsP[32][PSTRIDE];  // P^T tile, [m][256 n]
    __shared__ float smax[8][32];                       // per-wave per-m max
    __shared__ float psum[8][32];                       // per-wave per-m sum

    const int b    = blockIdx.x >> 7;           // N/32 m-tiles per batch
    const int mt   = (blockIdx.x & 127) * 32;
    const int wave = threadIdx.x >> 5;
    const int lane = threadIdx.x & 31;
    const int half = lane >> 4;
    const int l16  = lane & 15;
    const int c0   = wave * 32;                 // this wave's channel slice
    const int d0   = half * 16;                 // contiguous A-frag chunks

    // Loop-invariant B fragments: Q^T [32 d x 16 m] for both m-subtiles.
    BF16Frag qb0, qb1;
    {
        const __bf16* qrow = attnQ + (size_t)(b * N_DIM + mt + l16) * D_DIM + half * 16;
        qb0.h[0] = *(const v8bf*)(qrow);
        qb0.h[1] = *(const v8bf*)(qrow + 8);
        const __bf16* qrow1 = qrow + 16 * D_DIM;
        qb1.h[0] = *(const v8bf*)(qrow1);
        qb1.h[1] = *(const v8bf*)(qrow1 + 8);
    }

    v8f O00 = {}, O01 = {}, O10 = {}, O11 = {}; // [c-subtile][m-subtile]
    float mrun0 = -INFINITY, lrun0 = 0.0f;      // per-lane stats, m = mt+l16
    float mrun1 = -INFINITY, lrun1 = 0.0f;      // per-lane stats, m = mt+16+l16
    const float gamma = gammaPtr[0];

    for (int ns = 0; ns < N_DIM; ns += 256) {
        // ---- pass A: S^T tiles for this wave's 32 keys x 32 m ------------
        const int nblk = ns + wave * 32;
        BF16Frag ka0, ka1;                      // A frags: K rows n, k = d
        const __bf16* krow0 = attnK + (size_t)(b * N_DIM + nblk + l16) * D_DIM;
        ka0.h[0] = *(const v8bf*)(krow0 + d0);
        ka0.h[1] = *(const v8bf*)(krow0 + d0 + 8);
        const __bf16* krow1 = krow0 + 16 * D_DIM;
        ka1.h[0] = *(const v8bf*)(krow1 + d0);
        ka1.h[1] = *(const v8bf*)(krow1 + d0 + 8);

        v8f zero = {};
        v8f s00 = wmma_bf16(ka0.v, qb0.v, zero);   // [n-subtile][m-subtile]
        v8f s01 = wmma_bf16(ka0.v, qb1.v, zero);
        v8f s10 = wmma_bf16(ka1.v, qb0.v, zero);
        v8f s11 = wmma_bf16(ka1.v, qb1.v, zero);

        float mx0 = fmaxf(s00[0], s10[0]);
        float mx1 = fmaxf(s01[0], s11[0]);
        #pragma unroll
        for (int j = 1; j < 8; ++j) {
            mx0 = fmaxf(mx0, fmaxf(s00[j], s10[j]));
            mx1 = fmaxf(mx1, fmaxf(s01[j], s11[j]));
        }
        mx0 = fmaxf(mx0, __shfl_xor(mx0, 16));  // combine lane halves
        mx1 = fmaxf(mx1, __shfl_xor(mx1, 16));
        smax[wave][l16]      = mx0;             // both halves write same value
        smax[wave][16 + l16] = mx1;
        __syncthreads();

        // ---- pass B: global max, exp, publish P^T ------------------------
        float mnew0 = mrun0, mnew1 = mrun1;
        #pragma unroll
        for (int w = 0; w < 8; ++w) {
            mnew0 = fmaxf(mnew0, smax[w][l16]);
            mnew1 = fmaxf(mnew1, smax[w][16 + l16]);
        }
        const float sc0 = __expf(mrun0 - mnew0);   // 0 on first super-iter
        const float sc1 = __expf(mrun1 - mnew1);
        mrun0 = mnew0; mrun1 = mnew1;

        float ps0 = 0.0f, ps1 = 0.0f;
        #pragma unroll
        for (int j = 0; j < 8; ++j) {
            const float p00 = __expf(s00[j] - mnew0);
            const float p10 = __expf(s10[j] - mnew0);
            const float p01 = __expf(s01[j] - mnew1);
            const float p11 = __expf(s11[j] - mnew1);
            ps0 += p00 + p10;
            ps1 += p01 + p11;
            ldsP[l16][wave * 32 + half * 8 + j]           = (__bf16)p00;
            ldsP[l16][wave * 32 + 16 + half * 8 + j]      = (__bf16)p10;
            ldsP[16 + l16][wave * 32 + half * 8 + j]      = (__bf16)p01;
            ldsP[16 + l16][wave * 32 + 16 + half * 8 + j] = (__bf16)p11;
        }
        ps0 += __shfl_xor(ps0, 16);
        ps1 += __shfl_xor(ps1, 16);
        psum[wave][l16]      = ps0;
        psum[wave][16 + l16] = ps1;
        __syncthreads();

        float pst0 = 0.0f, pst1 = 0.0f;
        #pragma unroll
        for (int w = 0; w < 8; ++w) {
            pst0 += psum[w][l16];
            pst1 += psum[w][16 + l16];
        }
        lrun0 = lrun0 * sc0 + pst0;
        lrun1 = lrun1 * sc1 + pst1;
        #pragma unroll
        for (int j = 0; j < 8; ++j) {
            O00[j] *= sc0; O10[j] *= sc0;
            O01[j] *= sc1; O11[j] *= sc1;
        }

        // ---- pass C: O^T += V^T * P^T over all 256 keys ------------------
        for (int w2 = 0; w2 < 8; ++w2) {
            BF16Frag pb0, pb1;                  // B frags: P^T, lane col m
            const __bf16* prow0 = &ldsP[l16][w2 * 32 + half * 16];
            pb0.h[0] = *(const v8bf*)(prow0);
            pb0.h[1] = *(const v8bf*)(prow0 + 8);
            const __bf16* prow1 = &ldsP[16 + l16][w2 * 32 + half * 16];
            pb1.h[0] = *(const v8bf*)(prow1);
            pb1.h[1] = *(const v8bf*)(prow1 + 8);

            const int n0 = ns + w2 * 32;
            BF16Frag va0, va1;                  // A frags: V^T rows c, k = n
            const __bf16* vrow0 = attnV + (size_t)(b * C_DIM + c0 + l16) * N_DIM + n0;
            va0.h[0] = *(const v8bf*)(vrow0 + d0);
            va0.h[1] = *(const v8bf*)(vrow0 + d0 + 8);
            const __bf16* vrow1 = vrow0 + 16 * N_DIM;
            va1.h[0] = *(const v8bf*)(vrow1 + d0);
            va1.h[1] = *(const v8bf*)(vrow1 + d0 + 8);

            O00 = wmma_bf16(va0.v, pb0.v, O00);   // 4 WMMAs per V pair
            O01 = wmma_bf16(va0.v, pb1.v, O01);
            O10 = wmma_bf16(va1.v, pb0.v, O10);
            O11 = wmma_bf16(va1.v, pb1.v, O11);
        }
        // No extra barrier: next iteration's ldsP writes sit after its own
        // __syncthreads(), which all waves reach only after these reads.
    }

    // ---- epilogue: normalize (per-lane stats) and fused residual store ---
    const float inv0 = 1.0f / lrun0;
    const float inv1 = 1.0f / lrun1;
    #pragma unroll
    for (int j = 0; j < 8; ++j) {
        const int m0 = mt + l16;
        const int m1 = mt + 16 + l16;
        const size_t rA = (size_t)(b * C_DIM + c0 + j + half * 8) * N_DIM;
        const size_t rB = (size_t)(b * C_DIM + c0 + 16 + j + half * 8) * N_DIM;
        out[rA + m0] = x[rA + m0] + gamma * (O00[j] * inv0);  // 64B segments
        out[rA + m1] = x[rA + m1] + gamma * (O01[j] * inv1);
        out[rB + m0] = x[rB + m0] + gamma * (O10[j] * inv0);
        out[rB + m1] = x[rB + m1] + gamma * (O11[j] * inv1);
    }
}

// ---------------------------------------------------------------------------
// Launch: d_in = {x, Wq, bq, Wk, bk, Wv, bv, gamma}; d_out fp32 [B,C,H,W].
// Workspace layout (bytes):
//   attnK 1MB @0 | attnQ 1MB @1MB | attnV 8MB @2MB | xT 8MB @10MB
//   Wqb 16KB @18MB | Wkb 16KB @18MB+16K | Wvb 128KB @18MB+32K   (~18.2 MB)
// ---------------------------------------------------------------------------
extern "C" void kernel_launch(void* const* d_in, const int* in_sizes, int n_in,
                              void* d_out, int out_size, void* d_ws, size_t ws_size,
                              hipStream_t stream) {
    const float* x     = (const float*)d_in[0];
    const float* Wq    = (const float*)d_in[1];
    const float* bq    = (const float*)d_in[2];
    const float* Wk    = (const float*)d_in[3];
    const float* bk    = (const float*)d_in[4];
    const float* Wv    = (const float*)d_in[5];
    const float* bv    = (const float*)d_in[6];
    const float* gamma = (const float*)d_in[7];
    float* out = (float*)d_out;

    char* ws = (char*)d_ws;
    __bf16* attnK = (__bf16*)(ws);                             // [B][N][32]
    __bf16* attnQ = (__bf16*)(ws + (1u  << 20));               // [B][N][32]
    __bf16* attnV = (__bf16*)(ws + (2u  << 20));               // [B][C][N]
    __bf16* xT    = (__bf16*)(ws + (10u << 20));               // [B][N][C]
    __bf16* Wqb   = (__bf16*)(ws + (18u << 20));               // [32][256]
    __bf16* Wkb   = (__bf16*)(ws + (18u << 20) + (16u << 10)); // [32][256]
    __bf16* Wvb   = (__bf16*)(ws + (18u << 20) + (32u << 10)); // [256][256]

    transpose_x_kernel<<<B_DIM * 256, 256, 0, stream>>>(x, xT);
    convert_w_kernel<<<C_DIM, 256, 0, stream>>>(Wq, Wk, Wv, Wqb, Wkb, Wvb);

    proj_qk_kernel<<<B_DIM * (N_DIM / 128), 256, 0, stream>>>(
        xT, Wqb, bq, Wkb, bk, attnK, attnQ);
    proj_v_kernel<<<B_DIM * (N_DIM / 32), 256, 0, stream>>>(
        xT, Wvb, bv, attnV);

    sagan_flash_attn_kernel<<<B_DIM * (N_DIM / 32), 256, 0, stream>>>(
        x, gamma, attnQ, attnK, attnV, out);
}
